// InvariantPointAttention_26783416058540
// MI455X (gfx1250) — compile-verified, matrix operand order checked by reading, not run on hardware
//
#include <hip/hip_runtime.h>
#include <math.h>

constexpr int kH   = 12;
constexpr int kD   = 16;
constexpr int kPQK = 4;
constexpr int kPV  = 8;
constexpr int kCS  = 384;
constexpr int kCZ  = 128;
constexpr int kL   = 512;
constexpr int kNP  = 1152;   // 192*3 + 144*2 + 288 (q,k,v,qp,kp,vp concatenated)
constexpr int kVW  = 48;     // padded V width (40 used: 16 scalar + 24 point)

typedef __attribute__((ext_vector_type(16))) __bf16          v16bf;
typedef __attribute__((ext_vector_type(8)))  float           v8f;
typedef __attribute__((ext_vector_type(8)))  unsigned short  ushort8;
typedef __attribute__((ext_vector_type(16))) unsigned short  ushort16;
typedef __attribute__((ext_vector_type(4)))  unsigned int    uint4v;
typedef __attribute__((ext_vector_type(8)))  int             int8v;
typedef __attribute__((ext_vector_type(4)))  int             int4v;

__device__ __forceinline__ unsigned short f2bf(float f) {
  unsigned u = __builtin_bit_cast(unsigned, f);
  unsigned r = ((u >> 16) & 1u) + 0x7FFFu;           // round-to-nearest-even
  return (unsigned short)((u + r) >> 16);
}

// Load a 16x32 bf16 A/B fragment row-slice for this lane.
// lanes 0-15: row = lane, K = {0..7, 16..23}; lanes 16-31: row = lane-16, K = {8..15, 24..31}
// caller passes p = base + row*ld + k0 + klo  (klo = (lane>>4)*8)
__device__ __forceinline__ v16bf load_frag(const unsigned short* p) {
  ushort8 lo = *(const ushort8*)(p);
  ushort8 hi = *(const ushort8*)(p + 16);
  ushort16 v;
#pragma unroll
  for (int i = 0; i < 8; ++i) { v[i] = lo[i]; v[i + 8] = hi[i]; }
  return __builtin_bit_cast(v16bf, v);
}

// ---------------- 1) LayerNorm(s) -> bf16 ----------------
__global__ void k_ln_s(const float* __restrict__ s, const float* __restrict__ g,
                       const float* __restrict__ b, unsigned short* __restrict__ snb) {
  int wave = threadIdx.x >> 5, lane = threadIdx.x & 31;
  int row = blockIdx.x * 8 + wave;
  const float* x = s + (size_t)row * kCS;
  float v[12]; float sum = 0.f;
#pragma unroll
  for (int k = 0; k < 12; ++k) { v[k] = x[lane + 32 * k]; sum += v[k]; }
  for (int m = 16; m; m >>= 1) sum += __shfl_xor(sum, m);
  float mu = sum / kCS;
  float var = 0.f;
#pragma unroll
  for (int k = 0; k < 12; ++k) { float d = v[k] - mu; var += d * d; }
  for (int m = 16; m; m >>= 1) var += __shfl_xor(var, m);
  float rinv = rsqrtf(var / kCS + 1e-5f);
#pragma unroll
  for (int k = 0; k < 12; ++k) {
    int c = lane + 32 * k;
    snb[(size_t)row * kCS + c] = f2bf((v[k] - mu) * rinv * g[c] + b[c]);
  }
}

// ---------------- 2) pack all projection weights transposed -> bf16 [n][k] ----------------
__global__ void k_pack_w(const float* __restrict__ Wq, const float* __restrict__ Wk,
                         const float* __restrict__ Wv, const float* __restrict__ Wqp,
                         const float* __restrict__ Wkp, const float* __restrict__ Wvp,
                         unsigned short* __restrict__ Wt) {
  int tid = blockIdx.x * blockDim.x + threadIdx.x;
  if (tid >= kNP * kCS) return;
  int n = tid / kCS, k = tid % kCS;
  const float* W; int off, w;
  if      (n < 192) { W = Wq;  off = 0;   w = 192; }
  else if (n < 384) { W = Wk;  off = 192; w = 192; }
  else if (n < 576) { W = Wv;  off = 384; w = 192; }
  else if (n < 720) { W = Wqp; off = 576; w = 144; }
  else if (n < 864) { W = Wkp; off = 720; w = 144; }
  else              { W = Wvp; off = 864; w = 288; }
  Wt[(size_t)n * kCS + k] = f2bf(W[(size_t)k * w + (n - off)]);
}

// ---------------- 3) pair path: LN(z) @ Wpb -> logits (bandwidth bound, 134 MB stream) ----------------
__global__ void k_pair(const float* __restrict__ z, const float* __restrict__ gz,
                       const float* __restrict__ bz, const float* __restrict__ Wpb,
                       float* __restrict__ logits) {
  int wave = threadIdx.x >> 5, lane = threadIdx.x & 31;
  int pidx = blockIdx.x * 8 + wave;                 // (i,j) pair
  int i = pidx >> 9, j = pidx & 511;
  const float4 zv = *(const float4*)(z + (size_t)pidx * kCZ + lane * 4);
  __builtin_prefetch(z + (size_t)(pidx + 8) * kCZ + lane * 4, 0, 1); // global_prefetch_b8
  float sum = zv.x + zv.y + zv.z + zv.w;
  for (int m = 16; m; m >>= 1) sum += __shfl_xor(sum, m);
  float mu = sum / kCZ;
  float d0 = zv.x - mu, d1 = zv.y - mu, d2 = zv.z - mu, d3 = zv.w - mu;
  float var = d0 * d0 + d1 * d1 + d2 * d2 + d3 * d3;
  for (int m = 16; m; m >>= 1) var += __shfl_xor(var, m);
  float rinv = rsqrtf(var / kCZ + 1e-5f);
  int c0 = lane * 4;
  float xn[4] = { d0 * rinv * gz[c0]     + bz[c0],
                  d1 * rinv * gz[c0 + 1] + bz[c0 + 1],
                  d2 * rinv * gz[c0 + 2] + bz[c0 + 2],
                  d3 * rinv * gz[c0 + 3] + bz[c0 + 3] };
  float p[kH];
#pragma unroll
  for (int h = 0; h < kH; ++h)
    p[h] = xn[0] * Wpb[(c0)     * kH + h] + xn[1] * Wpb[(c0 + 1) * kH + h]
         + xn[2] * Wpb[(c0 + 2) * kH + h] + xn[3] * Wpb[(c0 + 3) * kH + h];
#pragma unroll
  for (int h = 0; h < kH; ++h) {
    float v = p[h];
    for (int m = 16; m; m >>= 1) v += __shfl_xor(v, m);
    if (lane == h) logits[((size_t)h * kL + i) * kL + j] = v;
  }
}

// ---------------- 4) QKV projections: sn(512x384) @ W(384x1152), WMMA bf16 ----------------
__global__ void k_proj(const unsigned short* __restrict__ snb,
                       const unsigned short* __restrict__ Wt,
                       float* __restrict__ proj) {
  int nt = blockIdx.x % (kNP / 16), mt = blockIdx.x / (kNP / 16);
  int lane = threadIdx.x;
  int row = lane & 15, hi = lane >> 4, klo = hi * 8;
  int m0 = mt * 16, n0 = nt * 16;
  v8f acc = {};
  for (int k0 = 0; k0 < kCS; k0 += 32) {
    v16bf a = load_frag(snb + (size_t)(m0 + row) * kCS + k0 + klo);
    v16bf b = load_frag(Wt  + (size_t)(n0 + row) * kCS + k0 + klo);
    acc = __builtin_amdgcn_wmma_f32_16x16x32_bf16(false, a, false, b, (short)0, acc, false, false);
  }
  int col = lane & 15, rs = hi * 8;
#pragma unroll
  for (int r = 0; r < 8; ++r)
    proj[(size_t)(m0 + rs + r) * kNP + n0 + col] = acc[r];
}

// ---------------- 5) build fused attention features ----------------
// A[i] = [q_s/sqrt(D) (16) | R q_pl + t (12) | 0 (4)]
// B[j] = [k_s (16)         | w_h*(R k_pl+t)  | 0 (4)]   => A.B = qk/4 + w*qp.kp
// rbias = -w/2*|q_p|^2 , cbias = -w/2*|k_p|^2 ;  Vt = [v_s | v_p] transposed, bf16
__global__ void k_build(const float* __restrict__ proj, const float* __restrict__ R,
                        const float* __restrict__ t, const float* __restrict__ pw,
                        unsigned short* __restrict__ Af, unsigned short* __restrict__ Bf,
                        unsigned short* __restrict__ Vt, float* __restrict__ rbias,
                        float* __restrict__ cbias) {
  int tid = blockIdx.x * blockDim.x + threadIdx.x;
  if (tid >= kH * kL) return;
  int h = tid / kL, l = tid % kL;
  const float* pr = proj + (size_t)l * kNP;
  const float* Rl = R + l * 9;
  float tx = t[l * 3 + 0], ty = t[l * 3 + 1], tz = t[l * 3 + 2];
  float pwv = pw[h];
  float w = (pwv > 20.f) ? pwv : log1pf(__expf(pwv));   // softplus
  unsigned short* A  = Af + ((size_t)h * kL + l) * 32;
  unsigned short* Bk = Bf + ((size_t)h * kL + l) * 32;
#pragma unroll
  for (int d = 0; d < 16; ++d) {
    A[d]  = f2bf(pr[h * 16 + d] * 0.25f);               // 1/sqrt(D)
    Bk[d] = f2bf(pr[192 + h * 16 + d]);
    Vt[((size_t)(h * kVW + d)) * kL + l] = f2bf(pr[384 + h * 16 + d]);
  }
  float qn2 = 0.f, kn2 = 0.f;
#pragma unroll
  for (int pp = 0; pp < kPQK; ++pp) {
    const float* q0 = pr + 576 + h * (kPQK * 3) + pp * 3;
    float qx = Rl[0] * q0[0] + Rl[1] * q0[1] + Rl[2] * q0[2] + tx;
    float qy = Rl[3] * q0[0] + Rl[4] * q0[1] + Rl[5] * q0[2] + ty;
    float qz = Rl[6] * q0[0] + Rl[7] * q0[1] + Rl[8] * q0[2] + tz;
    qn2 += qx * qx + qy * qy + qz * qz;
    A[16 + pp * 3 + 0] = f2bf(qx); A[16 + pp * 3 + 1] = f2bf(qy); A[16 + pp * 3 + 2] = f2bf(qz);
    const float* kk = pr + 720 + h * (kPQK * 3) + pp * 3;
    float kx = Rl[0] * kk[0] + Rl[1] * kk[1] + Rl[2] * kk[2] + tx;
    float ky = Rl[3] * kk[0] + Rl[4] * kk[1] + Rl[5] * kk[2] + ty;
    float kz = Rl[6] * kk[0] + Rl[7] * kk[1] + Rl[8] * kk[2] + tz;
    kn2 += kx * kx + ky * ky + kz * kz;
    Bk[16 + pp * 3 + 0] = f2bf(w * kx); Bk[16 + pp * 3 + 1] = f2bf(w * ky); Bk[16 + pp * 3 + 2] = f2bf(w * kz);
  }
#pragma unroll
  for (int c = 28; c < 32; ++c) { A[c] = 0; Bk[c] = 0; }
  rbias[h * kL + l] = -0.5f * w * qn2;
  cbias[h * kL + l] = -0.5f * w * kn2;
#pragma unroll
  for (int pp = 0; pp < kPV; ++pp) {
    const float* v0 = pr + 864 + h * (kPV * 3) + pp * 3;
    float vx = Rl[0] * v0[0] + Rl[1] * v0[1] + Rl[2] * v0[2] + tx;
    float vy = Rl[3] * v0[0] + Rl[4] * v0[1] + Rl[5] * v0[2] + ty;
    float vz = Rl[6] * v0[0] + Rl[7] * v0[1] + Rl[8] * v0[2] + tz;
    Vt[((size_t)(h * kVW + 16 + pp * 3 + 0)) * kL + l] = f2bf(vx);
    Vt[((size_t)(h * kVW + 16 + pp * 3 + 1)) * kL + l] = f2bf(vy);
    Vt[((size_t)(h * kVW + 16 + pp * 3 + 2)) * kL + l] = f2bf(vz);
  }
#pragma unroll
  for (int n = 40; n < kVW; ++n) Vt[((size_t)(h * kVW + n)) * kL + l] = 0;
}

// ---------------- 6) logits: one wave per (h, i-tile); A fragment reused across 32 j-tiles ----------------
__global__ void k_logits(const unsigned short* __restrict__ Af,
                         const unsigned short* __restrict__ Bf,
                         const float* __restrict__ rbias, const float* __restrict__ cbias,
                         float* __restrict__ logits) {
  int bid = blockIdx.x;                         // h*32 + it
  int h = bid >> 5, it = bid & 31;
  int lane = threadIdx.x;
  int row = lane & 15, hi = lane >> 4, klo = hi * 8;
  int i0 = it * 16;
  v16bf a = load_frag(Af + ((size_t)h * kL + i0 + row) * 32 + klo);
  int col = lane & 15, rs = hi * 8;
  float rbv[8];
#pragma unroll
  for (int r = 0; r < 8; ++r) rbv[r] = rbias[h * kL + i0 + rs + r];
  for (int jt = 0; jt < 32; ++jt) {
    int j0 = jt * 16;
    v16bf b = load_frag(Bf + ((size_t)h * kL + j0 + row) * 32 + klo);
    v8f acc = {};
    acc = __builtin_amdgcn_wmma_f32_16x16x32_bf16(false, a, false, b, (short)0, acc, false, false);
    float cbv = cbias[h * kL + j0 + col];
#pragma unroll
    for (int r = 0; r < 8; ++r) {
      size_t idx = ((size_t)h * kL + i0 + rs + r) * kL + j0 + col;
      logits[idx] = acc[r] + logits[idx] + rbv[r] + cbv;   // pair logits already there
    }
  }
}

// ---------------- 7) softmax over j, write bf16 probabilities ----------------
__global__ void k_softmax(const float* __restrict__ logits, unsigned short* __restrict__ attn) {
  __shared__ float red[128];
  int hi_idx = blockIdx.x;                      // h*L + i
  int tid = threadIdx.x;
  const float* row = logits + (size_t)hi_idx * kL;
  float4 v = *(const float4*)(row + tid * 4);
  float mx = fmaxf(fmaxf(v.x, v.y), fmaxf(v.z, v.w));
  red[tid] = mx; __syncthreads();
  for (int s = 64; s; s >>= 1) { if (tid < s) red[tid] = fmaxf(red[tid], red[tid + s]); __syncthreads(); }
  mx = red[0]; __syncthreads();
  float e0 = __expf(v.x - mx), e1 = __expf(v.y - mx), e2 = __expf(v.z - mx), e3 = __expf(v.w - mx);
  red[tid] = e0 + e1 + e2 + e3; __syncthreads();
  for (int s = 64; s; s >>= 1) { if (tid < s) red[tid] += red[tid + s]; __syncthreads(); }
  float rs = 1.f / red[0];
  unsigned short* o = attn + (size_t)hi_idx * kL + tid * 4;
  o[0] = f2bf(e0 * rs); o[1] = f2bf(e1 * rs); o[2] = f2bf(e2 * rs); o[3] = f2bf(e3 * rs);
}

// ---------------- 8) attn @ [v_s|v_p]: one WGP-block per head; Vt staged to LDS by the TDM ----------------
__global__ void __launch_bounds__(512) k_av(const unsigned short* __restrict__ attn,
                                            const unsigned short* __restrict__ Vt,
                                            float* __restrict__ outf) {
  __shared__ unsigned short vlds[kVW * kL];     // 48 x 512 bf16 = 48 KB
  int h = blockIdx.x;
  int wave = threadIdx.x >> 5, lane = threadIdx.x & 31;

  if (wave == 0) {
    // Tensor DMA descriptor (D#): 2D tile, 512 x 48 elements of 2 bytes, global -> LDS
    unsigned lds_base = (unsigned)(uintptr_t)(&vlds[0]);
    unsigned long long ga = (unsigned long long)(uintptr_t)(Vt + (size_t)h * kVW * kL);
    uint4v g0;
    g0[0] = 1u;                                            // count=1, user mode, no gather
    g0[1] = lds_base;                                      // lds_addr [63:32]
    g0[2] = (unsigned)(ga & 0xFFFFFFFFu);                  // global_addr[31:0]
    g0[3] = (unsigned)((ga >> 32) & 0x1FFFFFFu) | (2u << 30); // global_addr[56:32], type=2
    int8v g1;
    g1[0] = (int)(1u << 16);                               // wg_mask=0, data_size=1 (2B)
    g1[1] = (int)(((unsigned)kL & 0xFFFFu) << 16);         // tensor_dim0[15:0]=512 @bits63:48
    g1[2] = (int)(((unsigned)kVW) << 16);                  // tensor_dim1[15:0]=48  @bits95:80
    g1[3] = (int)(((unsigned)kL) << 16);                   // tile_dim0=512         @bits127:112
    g1[4] = (int)kVW;                                      // tile_dim1=48          @bits143:128
    g1[5] = (int)kL;                                       // tensor_dim0_stride=512 @bits191:160
    g1[6] = 0;                                             // stride[47:32]=0, dim1_stride=0
    g1[7] = 0;
    int4v g2 = {0, 0, 0, 0};                               // unused (2D tensor)
    int4v g3 = {0, 0, 0, 0};
    int8v g4 = {0, 0, 0, 0, 0, 0, 0, 0};                   // 6-arg toolchain extra group
    __builtin_amdgcn_tensor_load_to_lds(g0, g1, g2, g3, g4, 0);
    __builtin_amdgcn_s_wait_tensorcnt(0);
  }
  __syncthreads();

  int row = lane & 15, hi = lane >> 4, klo = hi * 8;
  int col = lane & 15, rs = hi * 8;
#pragma unroll
  for (int itpass = 0; itpass < 2; ++itpass) {
    int it = wave + itpass * 16;
    int i0 = it * 16;
    v8f acc0 = {}, acc1 = {}, acc2 = {};
    for (int kt = 0; kt < 16; ++kt) {
      int j0 = kt * 32;
      v16bf a  = load_frag(attn + ((size_t)h * kL + i0 + row) * kL + j0 + klo);
      v16bf b0 = load_frag(vlds + (size_t)(0  + row) * kL + j0 + klo);   // ds_load_b128
      v16bf b1 = load_frag(vlds + (size_t)(16 + row) * kL + j0 + klo);
      v16bf b2 = load_frag(vlds + (size_t)(32 + row) * kL + j0 + klo);
      acc0 = __builtin_amdgcn_wmma_f32_16x16x32_bf16(false, a, false, b0, (short)0, acc0, false, false);
      acc1 = __builtin_amdgcn_wmma_f32_16x16x32_bf16(false, a, false, b1, (short)0, acc1, false, false);
      acc2 = __builtin_amdgcn_wmma_f32_16x16x32_bf16(false, a, false, b2, (short)0, acc2, false, false);
    }
#pragma unroll
    for (int r = 0; r < 8; ++r) {
      float* o = outf + ((size_t)h * kL + i0 + rs + r) * kVW + col;
      o[0]  = acc0[r];
      o[16] = acc1[r];
      o[32] = acc2[r];
    }
  }
}

// ---------------- 9) inverse frame, norms, concat, output projection ----------------
__global__ void k_final(const float* __restrict__ outf, const float* __restrict__ R,
                        const float* __restrict__ t, const float* __restrict__ s,
                        const float* __restrict__ Wo, const float* __restrict__ bo,
                        float* __restrict__ out) {
  __shared__ float cat[576];
  int i = blockIdx.x, tid = threadIdx.x;
  if (tid < 192) {
    int h = tid >> 4, d = tid & 15;
    cat[tid] = outf[((size_t)h * kL + i) * kVW + d];
  } else if (tid < 288) {
    int hp = tid - 192;                         // 0..95 = (h,p)
    int h = hp >> 3, p = hp & 7;
    const float* o = outf + ((size_t)h * kL + i) * kVW + 16 + p * 3;
    float gx = o[0] - t[i * 3 + 0], gy = o[1] - t[i * 3 + 1], gz = o[2] - t[i * 3 + 2];
    const float* Rl = R + i * 9;
    float lx = Rl[0] * gx + Rl[3] * gy + Rl[6] * gz;   // R^T
    float ly = Rl[1] * gx + Rl[4] * gy + Rl[7] * gz;
    float lz = Rl[2] * gx + Rl[5] * gy + Rl[8] * gz;
    cat[192 + hp * 3 + 0] = lx; cat[192 + hp * 3 + 1] = ly; cat[192 + hp * 3 + 2] = lz;
    cat[480 + hp] = sqrtf(lx * lx + ly * ly + lz * lz + 1e-8f);
  }
  __syncthreads();
  float acc = bo[tid];
  for (int k = 0; k < 576; ++k) acc += cat[k] * Wo[(size_t)k * kCS + tid];
  out[(size_t)i * kCS + tid] = s[(size_t)i * kCS + tid] + acc;
}

extern "C" void kernel_launch(void* const* d_in, const int* in_sizes, int n_in,
                              void* d_out, int out_size, void* d_ws, size_t ws_size,
                              hipStream_t stream) {
  const float* s   = (const float*)d_in[0];
  const float* z   = (const float*)d_in[1];
  const float* R   = (const float*)d_in[2];
  const float* t   = (const float*)d_in[3];
  const float* g_s = (const float*)d_in[4];
  const float* b_s = (const float*)d_in[5];
  const float* g_z = (const float*)d_in[6];
  const float* b_z = (const float*)d_in[7];
  const float* Wq  = (const float*)d_in[8];
  const float* Wk  = (const float*)d_in[9];
  const float* Wv  = (const float*)d_in[10];
  const float* Wqp = (const float*)d_in[11];
  const float* Wkp = (const float*)d_in[12];
  const float* Wvp = (const float*)d_in[13];
  const float* Wpb = (const float*)d_in[14];
  const float* pw  = (const float*)d_in[15];
  const float* Wo  = (const float*)d_in[16];
  const float* bo  = (const float*)d_in[17];
  float* out = (float*)d_out;

  char* p = (char*)d_ws;
  auto alloc = [&](size_t bytes) { char* r = p; p += (bytes + 255) & ~(size_t)255; return r; };
  unsigned short* snb  = (unsigned short*)alloc((size_t)kL * kCS * 2);
  unsigned short* Wt   = (unsigned short*)alloc((size_t)kNP * kCS * 2);
  float*          proj = (float*)alloc((size_t)kL * kNP * 4);
  unsigned short* Af   = (unsigned short*)alloc((size_t)kH * kL * 32 * 2);
  unsigned short* Bf   = (unsigned short*)alloc((size_t)kH * kL * 32 * 2);
  unsigned short* Vt   = (unsigned short*)alloc((size_t)kH * kVW * kL * 2);
  float*          rb   = (float*)alloc((size_t)kH * kL * 4);
  float*          cb   = (float*)alloc((size_t)kH * kL * 4);
  float*          lg   = (float*)alloc((size_t)kH * kL * kL * 4);
  unsigned short* at   = (unsigned short*)alloc((size_t)kH * kL * kL * 2);
  float*          of   = (float*)alloc((size_t)kH * kL * kVW * 4);

  k_ln_s   <<<kL / 8, 256, 0, stream>>>(s, g_s, b_s, snb);
  k_pack_w <<<(kNP * kCS + 255) / 256, 256, 0, stream>>>(Wq, Wk, Wv, Wqp, Wkp, Wvp, Wt);
  k_pair   <<<kL * kL / 8, 256, 0, stream>>>(z, g_z, b_z, Wpb, lg);
  k_proj   <<<(kL / 16) * (kNP / 16), 32, 0, stream>>>(snb, Wt, proj);
  k_build  <<<(kH * kL + 255) / 256, 256, 0, stream>>>(proj, R, t, pw, Af, Bf, Vt, rb, cb);
  k_logits <<<kH * (kL / 16), 32, 0, stream>>>(Af, Bf, rb, cb, lg);
  k_softmax<<<kH * kL, 128, 0, stream>>>(lg, at);
  k_av     <<<kH, 512, 0, stream>>>(at, Vt, of);
  k_final  <<<kL, kCS, 0, stream>>>(of, R, t, s, Wo, bo, out);
}
